// RelNet_21895743275613
// MI455X (gfx1250) — matrix operand order repaired and checked by reading, not compiled, not run on hardware
//
#include <hip/hip_runtime.h>
#include <math.h>

// ---------------------------------------------------------------------------
// Types for CDNA5 WMMA (wave32): v_wmma_f32_16x16x32_bf16
// ---------------------------------------------------------------------------
typedef __attribute__((ext_vector_type(16))) __bf16 v16bf;
typedef __attribute__((ext_vector_type(8)))  float  v8f;

#define GEMM_MT 64
#define GEMM_NT 64
#define LDS_STRIDE 72   // 144 B row stride, multiple of 16 B

// ---------------------------------------------------------------------------
// Core bf16 WMMA GEMM:  Y(MxNc) = act( (W(MxKp) x X(KpxNc)) * scale[m] + shift[m] )
// W row-major bf16, X row-major bf16, Y row-major bf16.
// M % 64 == 0, Nc % 64 == 0, Kp % 64 == 0.
// Block: 256 threads = 8 wave32 waves; wave w -> m-tile (w&3), n-pair (w>>2).
// Each wave: 16x32 output = 2 f32 accumulators, 4 wmma per K-iteration.
// Staging: W and X tiles both loaded as 2x b128 per thread (coalesced),
// X transposed on the LDS-store side (16x ds_store_b16).
// ---------------------------------------------------------------------------
__global__ __launch_bounds__(256) void k_gemm_bf16(
    const __bf16* __restrict__ W, const __bf16* __restrict__ X,
    __bf16* __restrict__ Y,
    const float* __restrict__ scale, const float* __restrict__ shift,
    int M, int Kp, int Nc, int do_relu)
{
    __shared__ __align__(16) __bf16 sW[GEMM_MT][LDS_STRIDE];  // [m][k]
    __shared__ __align__(16) __bf16 sX[GEMM_NT][LDS_STRIDE];  // [col][k] (transposed)

    const int tid  = threadIdx.x;
    const int lane = tid & 31;
    const int wv   = tid >> 5;
    const int mt   = wv & 3;      // 4 m-tiles of 16
    const int nt   = wv >> 2;     // 2 n-pairs of 32
    const int hl   = lane >> 4;   // lane half (0/1)
    const int l15  = lane & 15;
    const int m0   = blockIdx.y * GEMM_MT;
    const int col0 = blockIdx.x * GEMM_NT;

    // cooperative-load maps (same shape for W and X tiles)
    const int wr  = tid >> 2;          // 0..63  row (W: m-row, X: k-row)
    const int wc  = (tid & 3) * 16;    // 0,16,32,48 (16 contiguous elements)

    // running global pointers (no per-element address math in the loop)
    const __bf16* wp = W + (size_t)(m0 + wr) * Kp + wc;
    const __bf16* xp = X + (size_t)wr * Nc + col0 + wc;
    const size_t  xstep = (size_t)64 * Nc;

    v8f acc0 = {}, acc1 = {};
    const int kt_n = Kp >> 6;
    for (int kt = 0; kt < kt_n; ++kt) {
        // --- stage W tile (64x64): 2x 16B vector loads, contiguous LDS store ---
        uint4 w0 = *reinterpret_cast<const uint4*>(wp);
        uint4 w1 = *reinterpret_cast<const uint4*>(wp + 8);
        *reinterpret_cast<uint4*>(&sW[wr][wc])     = w0;
        *reinterpret_cast<uint4*>(&sW[wr][wc + 8]) = w1;
        // --- stage X tile (64k x 64cols): 2x 16B vector loads of one k-row,
        //     transpose on the LDS-store side ---
        union XV { uint4 u[2]; __bf16 e[16]; } xv;
        xv.u[0] = *reinterpret_cast<const uint4*>(xp);
        xv.u[1] = *reinterpret_cast<const uint4*>(xp + 8);
        #pragma unroll
        for (int e = 0; e < 16; ++e) sX[wc + e][wr] = xv.e[e];

        wp += 64;
        xp += xstep;
        if (kt + 1 < kt_n) {  // gfx1250 global_prefetch_b8 for next tiles
            __builtin_prefetch(wp, 0, 0);
            __builtin_prefetch(xp, 0, 0);
        }
        __syncthreads();

        // --- A fragments for k-chunks 0/1: lane row (mt*16+l15),
        //     K = 16*half..16*half+15 within each 32-chunk ---
        union F { uint4 u[2]; v16bf v; };
        F a0, a1;
        const int arow = mt * 16 + l15;
        const int ab   = hl * 16;
        a0.u[0] = *reinterpret_cast<const uint4*>(&sW[arow][ab]);
        a0.u[1] = *reinterpret_cast<const uint4*>(&sW[arow][ab + 8]);
        a1.u[0] = *reinterpret_cast<const uint4*>(&sW[arow][32 + ab]);
        a1.u[1] = *reinterpret_cast<const uint4*>(&sW[arow][32 + ab + 8]);
        // --- B fragments: lane col, K = 16*half..16*half+15 per 32-chunk ---
        const int bb = hl * 16;
        {
            const int bcol = nt * 32 + l15;
            F b0, b1;
            b0.u[0] = *reinterpret_cast<const uint4*>(&sX[bcol][bb]);
            b0.u[1] = *reinterpret_cast<const uint4*>(&sX[bcol][bb + 8]);
            b1.u[0] = *reinterpret_cast<const uint4*>(&sX[bcol][32 + bb]);
            b1.u[1] = *reinterpret_cast<const uint4*>(&sX[bcol][32 + bb + 8]);
            acc0 = __builtin_amdgcn_wmma_f32_16x16x32_bf16(
                false, a0.v, false, b0.v, (short)0, acc0, false, false);
            acc0 = __builtin_amdgcn_wmma_f32_16x16x32_bf16(
                false, a1.v, false, b1.v, (short)0, acc0, false, false);
        }
        {
            const int bcol = nt * 32 + 16 + l15;
            F b0, b1;
            b0.u[0] = *reinterpret_cast<const uint4*>(&sX[bcol][bb]);
            b0.u[1] = *reinterpret_cast<const uint4*>(&sX[bcol][bb + 8]);
            b1.u[0] = *reinterpret_cast<const uint4*>(&sX[bcol][32 + bb]);
            b1.u[1] = *reinterpret_cast<const uint4*>(&sX[bcol][32 + bb + 8]);
            acc1 = __builtin_amdgcn_wmma_f32_16x16x32_bf16(
                false, a0.v, false, b0.v, (short)0, acc1, false, false);
            acc1 = __builtin_amdgcn_wmma_f32_16x16x32_bf16(
                false, a1.v, false, b1.v, (short)0, acc1, false, false);
        }
        __syncthreads();
    }

    // --- epilogue: D layout -> lane(l15)=col, vgpr r -> row r + 8*half ---
    #pragma unroll
    for (int s = 0; s < 2; ++s) {
        const int col = col0 + nt * 32 + s * 16 + l15;
        #pragma unroll
        for (int r = 0; r < 8; ++r) {
            const int m = m0 + mt * 16 + r + hl * 8;
            float v = (s == 0) ? acc0[r] : acc1[r];
            if (scale) v *= scale[m];
            if (shift) v += shift[m];
            if (do_relu && v < 0.f) v = 0.f;
            Y[(size_t)m * Nc + col] = (__bf16)v;
        }
    }
}

// ---------------------------------------------------------------------------
// im2col for 3x3 stride-2 pad-1 conv.  Output X is (Kp x N*S*S) bf16,
// row kk = ci*9 + ty*3 + tx (zero-padded for kk >= Cin*9).
// LAYOUT 0: input NCHW fp32 (the raw image)
// LAYOUT 1: input channel-major bf16: In[ci*(Nb*Sp*Sp) + n*Sp*Sp + iy*Sp + ix]
// ---------------------------------------------------------------------------
template <typename T, int LAYOUT>
__global__ void k_im2col(const T* __restrict__ In, __bf16* __restrict__ X,
                         int Nb, int Cin, int S, int Kp)
{
    const int Ncols = Nb * S * S;
    const int col = blockIdx.x * blockDim.x + threadIdx.x;
    const int kk  = blockIdx.y;
    if (col >= Ncols) return;
    const int Sp = S * 2;
    float val = 0.f;
    if (kk < Cin * 9) {
        const int ci = kk / 9;
        const int t  = kk - ci * 9;
        const int ty = t / 3, tx = t - ty * 3;
        const int n   = col / (S * S);
        const int rem = col - n * S * S;
        const int oy = rem / S, ox = rem - oy * S;
        const int iy = oy * 2 + ty - 1;
        const int ix = ox * 2 + tx - 1;
        if (iy >= 0 && iy < Sp && ix >= 0 && ix < Sp) {
            size_t idx;
            if (LAYOUT == 0)
                idx = (((size_t)n * Cin + ci) * Sp + iy) * Sp + ix;
            else
                idx = ((size_t)ci * Nb + n) * Sp * Sp + (size_t)iy * Sp + ix;
            val = (float)In[idx];
        }
    }
    X[(size_t)kk * Ncols + col] = (__bf16)val;
}

// fp32 -> bf16 row-slice convert with K padding (also plain convert when Kc==Kp)
__global__ void k_pad_rows(const float* __restrict__ src, __bf16* __restrict__ dst,
                           int M, int Ksrc, int off, int Kc, int Kp)
{
    const int i = blockIdx.x * blockDim.x + threadIdx.x;
    if (i >= M * Kp) return;
    const int m = i / Kp, k = i - m * Kp;
    const float v = (k < Kc) ? src[(size_t)m * Ksrc + off + k] : 0.f;
    dst[i] = (__bf16)v;
}

__global__ void k_zero_bf16(__bf16* __restrict__ p, int n)
{
    const int i = blockIdx.x * blockDim.x + threadIdx.x;
    if (i < n) p[i] = (__bf16)0.f;
}

// Fold conv bias + inference BatchNorm into per-channel scale/shift, 5 layers
__global__ void k_bn(const float* __restrict__ conv0_b, const float* __restrict__ convs_b,
                     const float* __restrict__ gamma, const float* __restrict__ beta,
                     const float* __restrict__ mu, const float* __restrict__ var,
                     float* __restrict__ scale, float* __restrict__ shift)
{
    const int i = blockIdx.x * blockDim.x + threadIdx.x;
    if (i >= 640) return;
    const int l = i >> 7, m = i & 127;
    const float s = gamma[i] * rsqrtf(var[i] + 1e-5f);
    const float cb = (l == 0) ? conv0_b[m] : convs_b[(l - 1) * 128 + m];
    scale[i] = s;
    shift[i] = (cb - mu[i]) * s + beta[i];
}

// One-layer LSTM over T=20 tokens, N=32 batches, E=HID=128.
// Block = 128 threads (one hidden unit each), grid = 32 (one batch each).
// Writes q transposed as bf16: qT[h*64 + n]  (GEMM "X", Kp=128, Nc=64 padded)
__global__ __launch_bounds__(128) void k_lstm(
    const int* __restrict__ ques, const float* __restrict__ emb,
    const float* __restrict__ Wih, const float* __restrict__ Whh,
    const float* __restrict__ bih, const float* __restrict__ bhh,
    __bf16* __restrict__ qT)
{
    __shared__ float sh[128], sx[128];
    const int t = threadIdx.x, n = blockIdx.x;
    sh[t] = 0.f;
    float c = 0.f;
    float bsum[4];
    #pragma unroll
    for (int g = 0; g < 4; ++g) bsum[g] = bih[g * 128 + t] + bhh[g * 128 + t];
    for (int step = 0; step < 20; ++step) {
        const int tok = ques[n * 20 + step];
        __syncthreads();
        sx[t] = emb[(size_t)tok * 128 + t];
        __syncthreads();
        float acc[4];
        #pragma unroll
        for (int g = 0; g < 4; ++g) acc[g] = bsum[g];
        for (int k = 0; k < 128; ++k) {
            const float xv = sx[k], hv = sh[k];
            #pragma unroll
            for (int g = 0; g < 4; ++g) {
                const int row = g * 128 + t;
                acc[g] += Wih[(size_t)row * 128 + k] * xv
                        + Whh[(size_t)row * 128 + k] * hv;
            }
        }
        const float ig = 1.f / (1.f + expf(-acc[0]));   // torch gate order i,f,g,o
        const float fg = 1.f / (1.f + expf(-acc[1]));
        const float gg = tanhf(acc[2]);
        const float og = 1.f / (1.f + expf(-acc[3]));
        c = fg * c + ig * gg;
        const float hnew = og * tanhf(c);
        __syncthreads();
        sh[t] = hnew;
    }
    __syncthreads();
    qT[t * 64 + n] = (__bf16)sh[t];
}

// Build padded object features (192 x 2048): rows 0..127 = conv4 channels,
// row 128 = o/8, row 129 = o%8, rows 130..191 = 0.  Column = n*64 + o.
__global__ void k_cells(const __bf16* __restrict__ Y4, __bf16* __restrict__ cells)
{
    const int i = blockIdx.x * blockDim.x + threadIdx.x;
    if (i >= 192 * 2048) return;
    const int r = i / 2048, col = i - r * 2048;
    float v;
    if (r < 128)       v = (float)Y4[r * 2048 + col];
    else if (r == 128) v = (float)((col & 63) >> 3);
    else if (r == 129) v = (float)((col & 63) & 7);
    else               v = 0.f;
    cells[i] = (__bf16)v;
}

// h1[c][n*4096 + i*64 + j] = relu(A[c][n*64+i] + B[c][n*64+j] + Qc[c][n])
__global__ __launch_bounds__(256) void k_assemble(
    const __bf16* __restrict__ A, const __bf16* __restrict__ B,
    const __bf16* __restrict__ Qc, __bf16* __restrict__ X1)
{
    const size_t idx = (size_t)blockIdx.x * blockDim.x + threadIdx.x; // 512*131072
    const int c  = (int)(idx >> 17);
    const int p  = (int)(idx & 131071);
    const int n  = p >> 12;
    const int ij = p & 4095;
    const int i  = ij >> 6, j = ij & 63;
    float v = (float)A[c * 2048 + n * 64 + i]
            + (float)B[c * 2048 + n * 64 + j]
            + (float)Qc[c * 64 + n];
    X1[idx] = (__bf16)(v > 0.f ? v : 0.f);
}

// Mean over the 4096 pairs: ctx[n][c] = mean_p H[c][n*4096+p].  One wave32 per (n,c).
__global__ __launch_bounds__(256) void k_meanpool(const __bf16* __restrict__ H,
                                                  float* __restrict__ ctx)
{
    const int wv = threadIdx.x >> 5, lane = threadIdx.x & 31;
    const int gw = blockIdx.x * 8 + wv;         // 0..16383
    const int n = gw >> 9, c = gw & 511;
    const __bf16* p = H + (size_t)c * 131072 + (size_t)n * 4096;
    float s = 0.f;
    for (int k = lane; k < 4096; k += 32) s += (float)p[k];
    #pragma unroll
    for (int m = 16; m >= 1; m >>= 1) s += __shfl_xor(s, m, 32);
    if (lane == 0) ctx[n * 512 + c] = s * (1.f / 4096.f);
}

// Small fp32 FC: out[n][m] = act(Xin[n]·Wt[m] + b[m])
__global__ void k_fc(const float* __restrict__ Xin, const float* __restrict__ Wt,
                     const float* __restrict__ b, float* __restrict__ out,
                     int Nr, int K, int Mo, int do_relu)
{
    const int i = blockIdx.x * blockDim.x + threadIdx.x;
    if (i >= Nr * Mo) return;
    const int n = i / Mo, m = i - n * Mo;
    float acc = b[m];
    const float* x = Xin + (size_t)n * K;
    const float* w = Wt + (size_t)m * K;
    for (int k = 0; k < K; ++k) acc += x[k] * w[k];
    if (do_relu && acc < 0.f) acc = 0.f;
    out[i] = acc;
}

// ---------------------------------------------------------------------------
// Host orchestration
// ---------------------------------------------------------------------------
extern "C" void kernel_launch(void* const* d_in, const int* in_sizes, int n_in,
                              void* d_out, int out_size, void* d_ws, size_t ws_size,
                              hipStream_t stream)
{
    (void)in_sizes; (void)n_in; (void)out_size; (void)ws_size;
    const float* img     = (const float*)d_in[0];
    const int*   ques    = (const int*)  d_in[1];
    const float* emb     = (const float*)d_in[2];
    const float* conv0_w = (const float*)d_in[3];
    const float* conv0_b = (const float*)d_in[4];
    const float* convs_w = (const float*)d_in[5];
    const float* convs_b = (const float*)d_in[6];
    const float* bn_g    = (const float*)d_in[7];
    const float* bn_b    = (const float*)d_in[8];
    const float* bn_m    = (const float*)d_in[9];
    const float* bn_v    = (const float*)d_in[10];
    const float* Wih     = (const float*)d_in[11];
    const float* Whh     = (const float*)d_in[12];
    const float* bih     = (const float*)d_in[13];
    const float* bhh     = (const float*)d_in[14];
    const float* g_w0    = (const float*)d_in[15];
    const float* g_b0    = (const float*)d_in[16];
    const float* g_w     = (const float*)d_in[17];
    const float* g_b     = (const float*)d_in[18];
    const float* f_w0    = (const float*)d_in[19];
    const float* f_b0    = (const float*)d_in[20];
    const float* f_w1    = (const float*)d_in[21];
    const float* f_b1    = (const float*)d_in[22];
    const float* f_w2    = (const float*)d_in[23];
    const float* f_b2    = (const float*)d_in[24];

    char* base = (char*)d_ws;
    size_t off = 0;
    auto take = [&](size_t bytes) -> char* {
        char* p = base + off;
        off = (off + bytes + 255) & ~(size_t)255;
        return p;
    };
    __bf16* wpConv0 = (__bf16*)take((size_t)128 * 64 * 2);
    __bf16* wpConvs = (__bf16*)take((size_t)4 * 128 * 1152 * 2);
    float*  bnScale = (float*) take(640 * 4);
    float*  bnShift = (float*) take(640 * 4);
    __bf16* wpW0L   = (__bf16*)take((size_t)512 * 192 * 2);
    __bf16* wpW0R   = (__bf16*)take((size_t)512 * 192 * 2);
    __bf16* wpW0Q   = (__bf16*)take((size_t)512 * 128 * 2);
    __bf16* wpG     = (__bf16*)take((size_t)3 * 512 * 512 * 2);
    __bf16* qT      = (__bf16*)take((size_t)128 * 64 * 2);
    __bf16* cells   = (__bf16*)take((size_t)192 * 2048 * 2);
    __bf16* Abuf    = (__bf16*)take((size_t)512 * 2048 * 2);
    __bf16* Bbuf    = (__bf16*)take((size_t)512 * 2048 * 2);
    __bf16* Qc      = (__bf16*)take((size_t)512 * 64 * 2);
    float*  ctx     = (float*) take((size_t)32 * 512 * 4);
    float*  z1      = (float*) take((size_t)32 * 512 * 4);
    float*  z2      = (float*) take((size_t)32 * 512 * 4);
    __bf16* R1      = (__bf16*)take((size_t)128 * 524288 * 2);   // conv Y / g pong
    __bf16* R2      = (__bf16*)take((size_t)1152 * 131072 * 2);  // im2col / g ping

    // ---- weight prep (fp32 -> bf16, K padding) ----
    k_pad_rows<<<(128 * 64 + 255) / 256, 256, 0, stream>>>(conv0_w, wpConv0, 128, 27, 0, 27, 64);
    k_pad_rows<<<(512 * 1152 + 255) / 256, 256, 0, stream>>>(convs_w, wpConvs, 512, 1152, 0, 1152, 1152);
    k_bn<<<3, 256, 0, stream>>>(conv0_b, convs_b, bn_g, bn_b, bn_m, bn_v, bnScale, bnShift);
    k_pad_rows<<<(512 * 192 + 255) / 256, 256, 0, stream>>>(g_w0, wpW0L, 512, 388, 0, 130, 192);
    k_pad_rows<<<(512 * 192 + 255) / 256, 256, 0, stream>>>(g_w0, wpW0R, 512, 388, 130, 130, 192);
    k_pad_rows<<<(512 * 128 + 255) / 256, 256, 0, stream>>>(g_w0, wpW0Q, 512, 388, 260, 128, 128);
    k_pad_rows<<<(1536 * 512 + 255) / 256, 256, 0, stream>>>(g_w, wpG, 1536, 512, 0, 512, 512);

    // ---- question LSTM (qT zero-padded to 64 batch columns) ----
    k_zero_bf16<<<(128 * 64 + 255) / 256, 256, 0, stream>>>(qT, 128 * 64);
    k_lstm<<<32, 128, 0, stream>>>(ques, emb, Wih, Whh, bih, bhh, qT);

    // ---- conv stack: im2col + WMMA GEMM with folded BN+ReLU ----
    k_im2col<float, 0><<<dim3(2048, 64), 256, 0, stream>>>(img, R2, 32, 3, 128, 64);
    k_gemm_bf16<<<dim3(524288 / 64, 2), 256, 0, stream>>>(wpConv0, R2, R1, bnScale, bnShift, 128, 64, 524288, 1);
    k_im2col<__bf16, 1><<<dim3(512, 1152), 256, 0, stream>>>(R1, R2, 32, 128, 64, 1152);
    k_gemm_bf16<<<dim3(131072 / 64, 2), 256, 0, stream>>>(wpConvs, R2, R1, bnScale + 128, bnShift + 128, 128, 1152, 131072, 1);
    k_im2col<__bf16, 1><<<dim3(128, 1152), 256, 0, stream>>>(R1, R2, 32, 128, 32, 1152);
    k_gemm_bf16<<<dim3(32768 / 64, 2), 256, 0, stream>>>(wpConvs + (size_t)1 * 128 * 1152, R2, R1, bnScale + 256, bnShift + 256, 128, 1152, 32768, 1);
    k_im2col<__bf16, 1><<<dim3(32, 1152), 256, 0, stream>>>(R1, R2, 32, 128, 16, 1152);
    k_gemm_bf16<<<dim3(8192 / 64, 2), 256, 0, stream>>>(wpConvs + (size_t)2 * 128 * 1152, R2, R1, bnScale + 384, bnShift + 384, 128, 1152, 8192, 1);
    k_im2col<__bf16, 1><<<dim3(8, 1152), 256, 0, stream>>>(R1, R2, 32, 128, 8, 1152);
    k_gemm_bf16<<<dim3(2048 / 64, 2), 256, 0, stream>>>(wpConvs + (size_t)3 * 128 * 1152, R2, R1, bnScale + 512, bnShift + 512, 128, 1152, 2048, 1);

    // ---- pair stage: g layer0 decomposed as per-object projections ----
    k_cells<<<(192 * 2048 + 255) / 256, 256, 0, stream>>>(R1, cells);
    k_gemm_bf16<<<dim3(2048 / 64, 8), 256, 0, stream>>>(wpW0L, cells, Abuf, nullptr, nullptr, 512, 192, 2048, 0);
    k_gemm_bf16<<<dim3(2048 / 64, 8), 256, 0, stream>>>(wpW0R, cells, Bbuf, nullptr, nullptr, 512, 192, 2048, 0);
    k_gemm_bf16<<<dim3(1, 8), 256, 0, stream>>>(wpW0Q, qT, Qc, nullptr, g_b0, 512, 128, 64, 0);
    k_assemble<<<262144, 256, 0, stream>>>(Abuf, Bbuf, Qc, R2);

    // ---- g layers 1..3 (512x512 WMMA GEMM over 131072 pair columns) ----
    k_gemm_bf16<<<dim3(2048, 8), 256, 0, stream>>>(wpG, R2, R1, nullptr, g_b, 512, 512, 131072, 1);
    k_gemm_bf16<<<dim3(2048, 8), 256, 0, stream>>>(wpG + (size_t)1 * 512 * 512, R1, R2, nullptr, g_b + 512, 512, 512, 131072, 1);
    k_gemm_bf16<<<dim3(2048, 8), 256, 0, stream>>>(wpG + (size_t)2 * 512 * 512, R2, R1, nullptr, g_b + 1024, 512, 512, 131072, 1);

    // ---- mean pool + f MLP ----
    k_meanpool<<<2048, 256, 0, stream>>>(R1, ctx);
    k_fc<<<64, 256, 0, stream>>>(ctx, f_w0, f_b0, z1, 32, 512, 512, 1);
    k_fc<<<64, 256, 0, stream>>>(z1, f_w1, f_b1, z2, 32, 512, 512, 1);
    k_fc<<<125, 256, 0, stream>>>(z2, f_w2, f_b2, (float*)d_out, 32, 512, 1000, 0);
}
